// ScaledHeadBlock_11287174054551
// MI455X (gfx1250) — matrix operand order repaired and checked by reading, not compiled, not run on hardware
//
#include <hip/hip_runtime.h>

// ---------------------------------------------------------------------------
// Types
// ---------------------------------------------------------------------------
typedef __bf16 bf16;
typedef __attribute__((ext_vector_type(16))) __bf16 v16bf;
typedef __attribute__((ext_vector_type(8)))  float  v8f;
typedef __attribute__((ext_vector_type(4)))  unsigned int v4u;
typedef __attribute__((ext_vector_type(4)))  int v4i;

union Frag16 {
    v16bf v;
    unsigned int u[8];
    __bf16 h[16];
};

union LdU {
    v4u v;
    __bf16 h[16];
};

// Async global->LDS copies (CDNA5 ASYNCcnt path), guarded so a toolchain
// without the builtins still compiles via the register-prefetch fallback.
#if defined(__HIP_DEVICE_COMPILE__) &&                                         \
    __has_builtin(__builtin_amdgcn_global_load_async_to_lds_b128) &&           \
    __has_builtin(__builtin_amdgcn_s_wait_asynccnt)
#define USE_ASYNC_LDS 1
#else
#define USE_ASYNC_LDS 0
#endif

#if USE_ASYNC_LDS
// builtin signature: (v4i addrspace(1)* src, v4i addrspace(3)* dst,
//                     imm int offset, imm int cpol)
typedef __attribute__((address_space(1))) v4i* as1_v4i_ptr;
typedef __attribute__((address_space(3))) v4i* as3_v4i_ptr;

static __device__ __forceinline__ void async_ld_b128(const void* g, void* l) {
    __builtin_amdgcn_global_load_async_to_lds_b128(
        (as1_v4i_ptr)(v4i*)g, (as3_v4i_ptr)(v4i*)l, 0, 0);
}
#endif

static __device__ __forceinline__ bf16 f2bf(float f) {
    // round-to-nearest-even f32 -> bf16
    unsigned int u = __builtin_bit_cast(unsigned int, f);
    unsigned int r = (u + 0x7FFFu + ((u >> 16) & 1u)) >> 16;
    unsigned short s = (unsigned short)r;
    return __builtin_bit_cast(bf16, s);
}

// ---------------------------------------------------------------------------
// Elementwise cast: f32 weights -> bf16
// ---------------------------------------------------------------------------
__global__ __launch_bounds__(256)
void cast_f32_to_bf16(const float* __restrict__ in, bf16* __restrict__ out, int n) {
    for (int i = blockIdx.x * 256 + threadIdx.x; i < n; i += gridDim.x * 256)
        out[i] = f2bf(in[i]);
}

// ---------------------------------------------------------------------------
// LayerNorm over last dim (C=1024), one block per row, output bf16
// ---------------------------------------------------------------------------
__global__ __launch_bounds__(256)
void layernorm_to_bf16(const float* __restrict__ x, const float* __restrict__ g,
                       const float* __restrict__ b, bf16* __restrict__ out, int C) {
    __shared__ float s1[256];
    __shared__ float s2[256];
    const int row = blockIdx.x;
    const int tid = threadIdx.x;
    const float* xr = x + (size_t)row * C;

    float lsum = 0.f, lsq = 0.f;
    for (int c = tid; c < C; c += 256) {
        float v = xr[c];
        lsum += v;
        lsq  += v * v;
    }
    s1[tid] = lsum; s2[tid] = lsq;
    __syncthreads();
    for (int s = 128; s > 0; s >>= 1) {
        if (tid < s) { s1[tid] += s1[tid + s]; s2[tid] += s2[tid + s]; }
        __syncthreads();
    }
    const float mu   = s1[0] / (float)C;
    const float var  = s2[0] / (float)C - mu * mu;
    const float rinv = rsqrtf(var + 1e-5f);

    bf16* orow = out + (size_t)row * C;
    for (int c = tid; c < C; c += 256)
        orow[c] = f2bf((xr[c] - mu) * rinv * g[c] + b[c]);
}

// ---------------------------------------------------------------------------
// Column-wise softmax (reference's axis=1).  One thread owns one column j and
// loops rows i -> fully coalesced row reads.  -inf entries contribute 0.
// ---------------------------------------------------------------------------
__global__ __launch_bounds__(256)
void softmax_columns(const float* __restrict__ S, bf16* __restrict__ P, int T) {
    const int j = blockIdx.x * 256 + threadIdx.x;
    const size_t base = (size_t)blockIdx.y * (size_t)T * (size_t)T;
    const float* s = S + base + j;
    bf16*       p = P + base + j;

    float mx = -__builtin_inff();
    for (int i = 0; i < T; ++i) mx = fmaxf(mx, s[(size_t)i * T]);
    float sum = 0.f;
    for (int i = 0; i < T; ++i) sum += __expf(s[(size_t)i * T] - mx);
    const float inv = 1.0f / sum;
    for (int i = 0; i < T; ++i)
        p[(size_t)i * T] = f2bf(__expf(s[(size_t)i * T] - mx) * inv);
}

// ---------------------------------------------------------------------------
// Tiled bf16 GEMM via v_wmma_f32_16x16x32_bf16, double-buffered LDS with the
// CDNA5 async global->LDS pipeline (fallback: register-prefetch pipeline).
//   C[m,n] = sum_k A[m,k] * B'[k,n]   (f32 accumulate)
//   TRANSB=1: B stored (N,K) row-major (weights / q^T);  staged as-is.
//   TRANSB=0: B stored (K,N) row-major (P @ v);  staged TRANSPOSED so LDS is
//             always N-major -> B fragment reads are contiguous b128s.
//   MASK : causal -inf where n>m; blocks fully above diagonal skip the K loop.
//   TRIK : A is lower-triangular (P) -> clamp K loop at m0+128.
// Tile: 128x128, K-step 32; 8 waves (4Mx2N), each wave 2x4 WMMA 16x16 tiles.
// ---------------------------------------------------------------------------
template <bool TRANSB, bool MASK, bool TRIK, bool BIAS, bool RELU, bool RES,
          bool OUTBF>
__global__ __launch_bounds__(256)
void gemm_bf16_wmma(const bf16* __restrict__ A, const bf16* __restrict__ Bm,
                    const float* __restrict__ bias, const float* __restrict__ res,
                    void* __restrict__ Cvoid,
                    int K, int lda, int ldb, int ldc,
                    size_t batchA, size_t batchB, size_t batchC, float scale) {
    __shared__ __bf16 As[2][128 * 32];
    __shared__ __bf16 Bs[2][128 * 32];  // always N-major: Bs[n*32 + k]

    const int z = blockIdx.z;
    A  += (size_t)z * batchA;
    Bm += (size_t)z * batchB;

    const int tid  = threadIdx.x;
    const int lane = tid & 31;
    const int wid  = tid >> 5;
    const int wm   = wid & 3;    // wave M slot -> +32 rows
    const int wn   = wid >> 2;   // wave N slot -> +64 cols
    const int l16  = lane & 15;
    const int hh   = lane >> 4;

    const int m0 = blockIdx.y * 128;
    const int n0 = blockIdx.x * 128;

    // Fully-masked score block: just emit -inf (softmax maps it to 0).
    if (MASK && n0 >= m0 + 128) {
        float* Cf = (float*)Cvoid + (size_t)z * batchC;
        for (int idx = tid; idx < 128 * 128; idx += 256) {
            const int rr = idx >> 7, cc = idx & 127;
            Cf[(size_t)(m0 + rr) * ldc + (n0 + cc)] = -__builtin_inff();
        }
        return;
    }

    int kend = K;
    if (TRIK) { const int lim = m0 + 128; kend = lim < K ? lim : K; }
    const int nT = kend >> 5;  // number of 32-wide K tiles

    // staging geometry: each thread moves 16 bf16 (32B) per matrix per tile
    const int sa_row = tid >> 1;          // 128 rows x 32 cols
    const int sa_col = (tid & 1) * 16;
    const int sbk = tid >> 3;             // !TRANSB source: 32(K) x 128(N)
    const int sbn = (tid & 7) * 16;

    v8f acc[2][4];
#pragma unroll
    for (int i = 0; i < 2; ++i)
#pragma unroll
        for (int j = 0; j < 4; ++j)
#pragma unroll
            for (int r = 0; r < 8; ++r) acc[i][j][r] = 0.0f;

    auto compute = [&](int buf) {
        // A frag 16x32: lane%16 = M; vgpr v -> K pair,
        // lane-half = K 0-15 / 16-31
        Frag16 af[2];
#pragma unroll
        for (int mi = 0; mi < 2; ++mi) {
            const int ar = wm * 32 + mi * 16 + l16;
#pragma unroll
            for (int v = 0; v < 8; ++v) {
                const int kk = (hh << 4) + (v << 1);
                af[mi].u[v] = *(const unsigned int*)(&As[buf][ar * 32 + kk]);
            }
        }
#pragma unroll
        for (int ni = 0; ni < 4; ++ni) {
            // B frag 32x16: lane%16 = N, lane-half = K 0-15 / 16-31
            Frag16 bfg;
            const int bn   = wn * 64 + ni * 16 + l16;
            const int koff = hh << 4;
#pragma unroll
            for (int v = 0; v < 8; ++v)
                bfg.u[v] = *(const unsigned int*)(&Bs[buf][bn * 32 + koff + (v << 1)]);
#pragma unroll
            for (int mi = 0; mi < 2; ++mi)
                acc[mi][ni] = __builtin_amdgcn_wmma_f32_16x16x32_bf16(
                    false, af[mi].v, false, bfg.v, (short)0, acc[mi][ni],
                    false, false);
        }
    };

#if USE_ASYNC_LDS
    if (TRANSB) {
        // async pipeline: issue tile t+1 into the other buffer while tile t
        // is consumed by the matrix pipe; s_wait_asynccnt 4 <=> tile t landed.
        auto issue = [&](int t, int buf) {
            const int k0 = t << 5;
            const bf16* ga = A  + (size_t)(m0 + sa_row) * lda + (k0 + sa_col);
            const bf16* gb = Bm + (size_t)(n0 + sa_row) * ldb + (k0 + sa_col);
            __bf16* la = &As[buf][tid * 16];
            __bf16* lb = &Bs[buf][tid * 16];
            async_ld_b128(ga, la);
            async_ld_b128(ga + 8, la + 8);
            async_ld_b128(gb, lb);
            async_ld_b128(gb + 8, lb + 8);
        };
        issue(0, 0);
        for (int t = 0; t < nT; ++t) {
            const int cur = t & 1;
            __syncthreads();  // nobody still reading buf[1-cur]
            if (t + 1 < nT) {
                issue(t + 1, 1 - cur);
                __builtin_amdgcn_s_wait_asynccnt(4);  // tile t (ours) done
            } else {
                __builtin_amdgcn_s_wait_asynccnt(0);
            }
            __syncthreads();  // tile t visible from every wave
            compute(cur);
        }
    } else
#endif
    {
        // fallback / transpose-staging pipeline: prefetch next tile into
        // registers while computing; one barrier per iteration.
        LdU a_ld, b_ld;
        auto gload = [&](int t) {
            const int k0 = t << 5;
            a_ld.v = *(const v4u*)(A + (size_t)(m0 + sa_row) * lda + (k0 + sa_col));
            if (TRANSB)
                b_ld.v = *(const v4u*)(Bm + (size_t)(n0 + sa_row) * ldb + (k0 + sa_col));
            else
                b_ld.v = *(const v4u*)(Bm + (size_t)(k0 + sbk) * ldb + (n0 + sbn));
        };
        gload(0);
        for (int t = 0; t < nT; ++t) {
            const int cur = t & 1;
            *(v4u*)(&As[cur][tid * 16]) = a_ld.v;
            if (TRANSB) {
                *(v4u*)(&Bs[cur][tid * 16]) = b_ld.v;
            } else {
                // transpose during staging -> N-major LDS, b128 frag reads
#pragma unroll
                for (int j = 0; j < 16; ++j)
                    Bs[cur][(sbn + j) * 32 + sbk] = b_ld.h[j];
            }
            __syncthreads();
            if (t + 1 < nT) gload(t + 1);
            if (t + 2 < nT)  // cache hint for the tile after next
                __builtin_prefetch(A + (size_t)(m0 + sa_row) * lda +
                                       ((t + 2) << 5) + sa_col, 0, 1);
            compute(cur);
        }
    }

    // Epilogue.  C/D layout: lane%16 = N, vgpr r -> M = r + 8*(lane/16)
#pragma unroll
    for (int mi = 0; mi < 2; ++mi) {
#pragma unroll
        for (int ni = 0; ni < 4; ++ni) {
            const int gn = n0 + wn * 64 + ni * 16 + l16;
#pragma unroll
            for (int r = 0; r < 8; ++r) {
                const int gm = m0 + wm * 32 + mi * 16 + (hh << 3) + r;
                float vv = acc[mi][ni][r] * scale;
                if (MASK && gn > gm) vv = -__builtin_inff();
                if (BIAS) vv += bias[gn];
                if (RELU) vv = fmaxf(vv, 0.0f);
                if (RES)  vv += res[(size_t)gm * ldc + gn];
                const size_t off = (size_t)z * batchC + (size_t)gm * ldc + gn;
                if (OUTBF) ((bf16*)Cvoid)[off] = f2bf(vv);
                else       ((float*)Cvoid)[off] = vv;
            }
        }
    }
}

// ---------------------------------------------------------------------------
// Host-side orchestration
// ---------------------------------------------------------------------------
extern "C" void kernel_launch(void* const* d_in, const int* in_sizes, int n_in,
                              void* d_out, int out_size, void* d_ws, size_t ws_size,
                              hipStream_t stream) {
    (void)in_sizes; (void)n_in; (void)out_size; (void)ws_size;

    const int Bn = 4, T = 2048, C = 1024;
    const int BT = Bn * T;                 // 8192 rows
    const int C2 = 2 * C, C4 = 4 * C;      // 2048, 4096
    const size_t TT = (size_t)T * T;
    const size_t TD = (size_t)T * C2;

    const float* x   = (const float*)d_in[0];
    const float* Wk  = (const float*)d_in[1];
    const float* Wq  = (const float*)d_in[2];
    const float* Wv  = (const float*)d_in[3];
    const float* Wl  = (const float*)d_in[4];
    const float* bl  = (const float*)d_in[5];
    const float* W1  = (const float*)d_in[6];
    const float* b1  = (const float*)d_in[7];
    const float* W2  = (const float*)d_in[8];
    const float* b2  = (const float*)d_in[9];
    const float* g1  = (const float*)d_in[10];
    const float* be1 = (const float*)d_in[11];
    const float* g2  = (const float*)d_in[12];
    const float* be2 = (const float*)d_in[13];

    char*  base = (char*)d_ws;
    size_t off  = 0;
    auto carve = [&](size_t bytes) -> char* {
        char* p = base + off;
        off = (off + bytes + 255) & ~(size_t)255;
        return p;
    };
    bf16*  h_bf   = (bf16*)carve((size_t)BT * C  * 2);
    bf16*  h2_bf  = (bf16*)carve((size_t)BT * C  * 2);
    bf16*  Wk_bf  = (bf16*)carve((size_t)C2 * C  * 2);
    bf16*  Wq_bf  = (bf16*)carve((size_t)C2 * C  * 2);
    bf16*  Wv_bf  = (bf16*)carve((size_t)C2 * C  * 2);
    bf16*  Wl_bf  = (bf16*)carve((size_t)C  * C2 * 2);
    bf16*  W1_bf  = (bf16*)carve((size_t)C4 * C  * 2);
    bf16*  W2_bf  = (bf16*)carve((size_t)C  * C4 * 2);
    bf16*  k_bf   = (bf16*)carve((size_t)BT * C2 * 2);
    bf16*  q_bf   = (bf16*)carve((size_t)BT * C2 * 2);
    bf16*  v_bf   = (bf16*)carve((size_t)BT * C2 * 2);
    float* S_f    = (float*)carve((size_t)Bn * TT * 4);
    bf16*  P_bf   = (bf16*)carve((size_t)Bn * TT * 2);
    bf16*  att_bf = (bf16*)carve((size_t)BT * C2 * 2);
    float* out_f  = (float*)carve((size_t)BT * C  * 4);
    bf16*  ff_bf  = (bf16*)carve((size_t)BT * C4 * 2);

    // 1) weights -> bf16
    cast_f32_to_bf16<<<2048, 256, 0, stream>>>(Wk, Wk_bf, C2 * C);
    cast_f32_to_bf16<<<2048, 256, 0, stream>>>(Wq, Wq_bf, C2 * C);
    cast_f32_to_bf16<<<2048, 256, 0, stream>>>(Wv, Wv_bf, C2 * C);
    cast_f32_to_bf16<<<2048, 256, 0, stream>>>(Wl, Wl_bf, C * C2);
    cast_f32_to_bf16<<<2048, 256, 0, stream>>>(W1, W1_bf, C4 * C);
    cast_f32_to_bf16<<<2048, 256, 0, stream>>>(W2, W2_bf, C * C4);

    // 2) h = LN1(x)
    layernorm_to_bf16<<<BT, 256, 0, stream>>>(x, g1, be1, h_bf, C);

    // 3) k/q/v = h @ W^T     (M=8192, N=2048, K=1024)
    dim3 gproj(C2 / 128, BT / 128, 1);
    gemm_bf16_wmma<true, false, false, false, false, false, true>
        <<<gproj, 256, 0, stream>>>(h_bf, Wk_bf, nullptr, nullptr, k_bf,
                                    C, C, C, C2, 0, 0, 0, 1.0f);
    gemm_bf16_wmma<true, false, false, false, false, false, true>
        <<<gproj, 256, 0, stream>>>(h_bf, Wq_bf, nullptr, nullptr, q_bf,
                                    C, C, C, C2, 0, 0, 0, 1.0f);
    gemm_bf16_wmma<true, false, false, false, false, false, true>
        <<<gproj, 256, 0, stream>>>(h_bf, Wv_bf, nullptr, nullptr, v_bf,
                                    C, C, C, C2, 0, 0, 0, 1.0f);

    // 4) S = (k @ q^T) * sqrt(C) with causal mask; above-diagonal blocks skip
    dim3 gattn(T / 128, T / 128, Bn);
    gemm_bf16_wmma<true, true, false, false, false, false, false>
        <<<gattn, 256, 0, stream>>>(k_bf, q_bf, nullptr, nullptr, S_f,
                                    C2, C2, C2, T, TD, TD, TT, 32.0f);

    // 5) P = softmax(S, axis=i)
    softmax_columns<<<dim3(T / 256, Bn), 256, 0, stream>>>(S_f, P_bf, T);

    // 6) att = P @ v  (B staged transposed; K clamped to lower triangle)
    gemm_bf16_wmma<false, false, true, false, false, false, true>
        <<<gattn, 256, 0, stream>>>(P_bf, v_bf, nullptr, nullptr, att_bf,
                                    T, T, C2, C2, TT, TD, TD, 1.0f);

    // 7) out = x + relu(att @ Wl^T + bl)
    gemm_bf16_wmma<true, false, false, true, true, true, false>
        <<<dim3(C / 128, BT / 128, 1), 256, 0, stream>>>(
            att_bf, Wl_bf, bl, x, out_f, C2, C2, C2, C, 0, 0, 0, 1.0f);

    // 8) h2 = LN2(out)
    layernorm_to_bf16<<<BT, 256, 0, stream>>>(out_f, g2, be2, h2_bf, C);

    // 9) ff = relu(h2 @ W1^T + b1)
    gemm_bf16_wmma<true, false, false, true, true, false, true>
        <<<dim3(C4 / 128, BT / 128, 1), 256, 0, stream>>>(
            h2_bf, W1_bf, b1, nullptr, ff_bf, C, C, C, C4, 0, 0, 0, 1.0f);

    // 10) final = out + (ff @ W2^T + b2) -> d_out
    gemm_bf16_wmma<true, false, false, true, false, true, false>
        <<<dim3(C / 128, BT / 128, 1), 256, 0, stream>>>(
            ff_bf, W2_bf, b2, out_f, (float*)d_out, C4, C4, C4, C, 0, 0, 0, 1.0f);
}
